// CasualSelfAttention_25744033972692
// MI455X (gfx1250) — matrix operand order repaired and checked by reading, not compiled, not run on hardware
//
#include <hip/hip_runtime.h>
#include <hip/hip_bf16.h>

// ---------------------------------------------------------------------------
// Causal self-attention, CDNA5 (gfx1250), wave32.
//   B=2, S=2048, D=1024, H=16, Dh=64
//
// Roofline: ~69 GFLOP total, ~100 MB HBM traffic (~4us @ 23.3 TB/s) ->
// matrix-pipe bound. bf16 WMMA (v_wmma_f32_16x16x32_bf16, fp32 accum) is the
// right precision/throughput point (fp32 WMMA is K=4, 8x more instructions).
//
// Operand tiles are staged through LDS with GLOBAL_LOAD_ASYNC_TO_LDS_B128
// double buffering (ASYNCcnt + barrier). The async asm takes the LDS address
// via ptrtoint of the real shared pointer so the shared array escapes and the
// compiler keeps/orders the LDS fragment reads (fixes round-2 undef folding).
// ---------------------------------------------------------------------------

typedef __bf16 bf16_t;
typedef __attribute__((ext_vector_type(16))) __bf16 v16bf;
typedef __attribute__((ext_vector_type(8)))  __bf16 v8bf;
typedef __attribute__((ext_vector_type(4)))  __bf16 v4bf;
typedef __attribute__((ext_vector_type(8)))  float  v8f;
typedef __attribute__((ext_vector_type(4)))  float  v4f;

#define BATCH   2
#define SEQ     2048
#define DEMB    1024
#define NHEAD   16
#define DHEAD   64
#define NTOK    (BATCH * SEQ)          // 4096
#define SCALE   0.125f                 // 1/sqrt(64)
#define NEG_BIG (-3.0e38f)

__device__ __forceinline__ v8f vzero8() {
    v8f z;
#pragma unroll
    for (int i = 0; i < 8; ++i) z[i] = 0.0f;
    return z;
}

__device__ __forceinline__ v8f wmma_bf16(v16bf a, v16bf b, v8f c) {
    return __builtin_amdgcn_wmma_f32_16x16x32_bf16(
        /*neg_a=*/false, a, /*neg_b=*/false, b,
        /*c_mod=*/(short)0, c, /*reuse_a=*/false, /*reuse_b=*/false);
}

union AFrag { v16bf v; v8bf h[2]; };

// --- async copy: 16 bytes per lane, global -> LDS (ASYNCcnt tracked) -------
// lds_ptr is a flat pointer into __shared__ memory. ptrtoint both yields the
// LDS byte offset (LDS aperture lives in the high 32 bits only) and captures
// the shared array so the "memory" clobber covers it.
__device__ __forceinline__ void async_b128(void* lds_ptr, const void* gptr) {
    const unsigned lds_off = (unsigned)(size_t)lds_ptr;
    asm volatile("global_load_async_to_lds_b128 %0, %1, off"
                 :: "v"(lds_off), "v"(gptr)
                 : "memory");
}

// drain this wave's async transfers, then workgroup barrier
__device__ __forceinline__ void wait_async_and_barrier() {
    asm volatile("s_wait_asynccnt 0x0" ::: "memory");
    __syncthreads();
}

// A-operand (16x32, M x K), row-major source with leading dim ld (elements).
// Layout used here (same k order as load_b_frag): lane L holds row (L&15), K = half*16 .. half*16+15 contiguous.
__device__ __forceinline__ v16bf load_a_frag(const bf16_t* base, int ld,
                                             int m0, int k0, int lane) {
    const int half = lane >> 4;
    const int row  = lane & 15;
    const bf16_t* p = base + (size_t)(m0 + row) * ld + k0;
    AFrag f;
    f.h[0] = *(const v8bf*)(p + half * 16);
    f.h[1] = *(const v8bf*)(p + half * 16 + 8);
    return f.v;
}

// B-operand (32x16, K x N): B(k,n) = src[n0+n][k0+k], source row-major [N,K].
// Layout used here (same k order as load_a_frag): lane L holds col (L&15), K = half*16 .. half*16+15 contiguous.
__device__ __forceinline__ v16bf load_b_frag(const bf16_t* base, int ld,
                                             int n0, int k0, int lane) {
    const int half = lane >> 4;
    const int col  = lane & 15;
    return *(const v16bf*)(base + (size_t)(n0 + col) * ld + k0 + half * 16);
}

// ---------------------------------------------------------------------------
// 1) fp32 -> bf16 conversion, 4 elements/thread
// ---------------------------------------------------------------------------
__global__ void cvt_f32_bf16(const float* __restrict__ src,
                             bf16_t* __restrict__ dst, int n) {
    int i = (blockIdx.x * blockDim.x + threadIdx.x) * 4;
    if (i >= n) return;
    v4f f = *(const v4f*)(src + i);
    v4bf b;
#pragma unroll
    for (int j = 0; j < 4; ++j) b[j] = (bf16_t)f[j];
    *(v4bf*)(dst + i) = b;
}

// ---------------------------------------------------------------------------
// Shared GEMM core: y = A @ W^T with LDS double-buffered async staging.
//   block = 256 thr (8 waves: 4 M-waves x 2 N-waves), tile 64M x 128N.
//   LDS: A[2][64x32] (4KB each) @ 0, B[2][128x32] (8KB each) @ 8192.
// ---------------------------------------------------------------------------
#define GEMM_LDS_BYTES (8192 + 16384)

__device__ __forceinline__ void gemm_issue_tiles(char* smem,
                                                 const bf16_t* gA,
                                                 const bf16_t* gB,
                                                 int k0, int buf, int tid) {
    {   // A tile 64x32: 256 chunks of 8 bf16 -> one per thread
        const int row = tid >> 2, cc = (tid & 3) * 8;
        async_b128(smem + buf * 4096 + (row * 32 + cc) * 2,
                   gA + (size_t)row * DEMB + k0 + cc);
    }
#pragma unroll
    for (int h = 0; h < 2; ++h) {   // B tile 128x32: 512 chunks -> two/thread
        const int c = tid + h * 256;
        const int row = c >> 2, cc = (c & 3) * 8;
        async_b128(smem + 8192 + buf * 8192 + (row * 32 + cc) * 2,
                   gB + (size_t)row * DEMB + k0 + cc);
    }
}

__device__ __forceinline__ void gemm_core(const bf16_t* gA, const bf16_t* gB,
                                          v8f acc[4], char* smem,
                                          int tid, int lane, int w) {
    const bf16_t* smA = (const bf16_t*)smem;
    const bf16_t* smB = (const bf16_t*)(smem + 8192);
    const int wm = (w & 3) * 16;
    const int wn = (w >> 2) * 64;

    gemm_issue_tiles(smem, gA, gB, 0, 0, tid);
    wait_async_and_barrier();

    for (int k0 = 0; k0 < DEMB; k0 += 32) {
        const int buf = (k0 >> 5) & 1;
        if (k0 + 32 < DEMB) gemm_issue_tiles(smem, gA, gB, k0 + 32, buf ^ 1, tid);

        v16bf a = load_a_frag(smA + buf * 2048, 32, wm, 0, lane);
#pragma unroll
        for (int t = 0; t < 4; ++t) {
            v16bf b = load_b_frag(smB + buf * 4096, 32, wn + 16 * t, 0, lane);
            acc[t] = wmma_bf16(a, b, acc[t]);
        }
        wait_async_and_barrier();
    }
}

// ---------------------------------------------------------------------------
// 2) QKV projections. grid = (N/128, M/64, 3).
//    z=0 -> Q [B,H,S,Dh]; z=1 -> K [B,H,S,Dh]; z=2 -> V^T [B,H,Dh,S]
// ---------------------------------------------------------------------------
__global__ __launch_bounds__(256)
void gemm_qkv(const bf16_t* __restrict__ xb,
              const bf16_t* __restrict__ wq,
              const bf16_t* __restrict__ wk,
              const bf16_t* __restrict__ wv,
              bf16_t* __restrict__ qo,
              bf16_t* __restrict__ ko,
              bf16_t* __restrict__ vo) {
    __shared__ __align__(32) char smem[GEMM_LDS_BYTES];
    const int tid  = threadIdx.x;
    const int lane = tid & 31;
    const int w    = tid >> 5;
    const int mblk = blockIdx.y * 64;
    const int nblk = blockIdx.x * 128;
    const int z    = blockIdx.z;
    const bf16_t* W = (z == 0) ? wq : (z == 1) ? wk : wv;

    v8f acc[4];
#pragma unroll
    for (int t = 0; t < 4; ++t) acc[t] = vzero8();

    gemm_core(xb + (size_t)mblk * DEMB, W + (size_t)nblk * DEMB,
              acc, smem, tid, lane, w);

    const int half = lane >> 4;
    const int idx  = lane & 15;
    const int m0   = mblk + (w & 3) * 16;
    const int n0   = nblk + (w >> 2) * 64;
#pragma unroll
    for (int t = 0; t < 4; ++t) {
#pragma unroll
        for (int r = 0; r < 8; ++r) {
            const int m  = m0 + r + 8 * half;        // token row
            const int n  = n0 + 16 * t + idx;        // embed col
            const int b  = m >> 11;
            const int s  = m & (SEQ - 1);
            const int h  = n >> 6;
            const int dh = n & (DHEAD - 1);
            const bf16_t val = (bf16_t)acc[t][r];
            if (z == 2) {
                vo[(((size_t)(b * NHEAD + h) * DHEAD + dh) * SEQ) + s] = val;
            } else {
                bf16_t* o = z ? ko : qo;
                o[(((size_t)(b * NHEAD + h) * SEQ + s) * DHEAD) + dh] = val;
            }
        }
    }
}

// ---------------------------------------------------------------------------
// 3) Flash attention. grid = (S/128, B*H), block = 256 (8 waves).
//    Block walks a UNIFORM key loop to its last needed key (qb0+127); K/V
//    32-key tiles are async-staged into LDS once per block (double buffered)
//    and shared by all 8 waves. Waves skip compute on fully-masked steps.
//    LDS: K[2][32x64] @0 (8KB), V[2][64x32] @8192 (8KB), P[8][16x32] @16384.
// ---------------------------------------------------------------------------
#define ATTN_LDS_BYTES (8192 + 8192 + 8192)

__device__ __forceinline__ void attn_issue_kv(char* smem,
                                              const bf16_t* K,
                                              const bf16_t* Vt,
                                              int j0, int buf, int tid) {
    {   // K tile 32 rows x 64 dh : 256 chunks
        const int row = tid >> 3, cc = (tid & 7) * 8;
        async_b128(smem + buf * 4096 + (row * 64 + cc) * 2,
                   K + (size_t)(j0 + row) * DHEAD + cc);
    }
    {   // V tile: Vt[dh=64][keys 32] : 256 chunks
        const int row = tid >> 2, cc = (tid & 3) * 8;
        async_b128(smem + 8192 + buf * 4096 + (row * 32 + cc) * 2,
                   Vt + (size_t)row * SEQ + j0 + cc);
    }
}

__global__ __launch_bounds__(256)
void flash_attn(const bf16_t* __restrict__ Qb,
                const bf16_t* __restrict__ Kb,
                const bf16_t* __restrict__ Vtb,
                bf16_t* __restrict__ Ob) {
    __shared__ __align__(32) char smem[ATTN_LDS_BYTES];
    const bf16_t* smK = (const bf16_t*)smem;
    const bf16_t* smV = (const bf16_t*)(smem + 8192);
    bf16_t*       pb  = (bf16_t*)(smem + 16384);

    const int tid  = threadIdx.x;
    const int lane = tid & 31;
    const int w    = tid >> 5;
    const int half = lane >> 4;
    const int idx  = lane & 15;
    const int bh   = blockIdx.y;
    const int qb0  = blockIdx.x * 128;      // block's first query row
    const int q0   = qb0 + w * 16;          // wave's first query row

    const bf16_t* Q  = Qb  + (size_t)bh * SEQ * DHEAD;
    const bf16_t* K  = Kb  + (size_t)bh * SEQ * DHEAD;
    const bf16_t* Vt = Vtb + (size_t)bh * DHEAD * SEQ;
    bf16_t*       pw = pb + w * (16 * 32);

    const v16bf aq0 = load_a_frag(Q, DHEAD, q0, 0,  lane);
    const v16bf aq1 = load_a_frag(Q, DHEAD, q0, 32, lane);

    v8f   o[4];
    float mi[8], li[8];
#pragma unroll
    for (int t = 0; t < 4; ++t) o[t] = vzero8();
#pragma unroll
    for (int r = 0; r < 8; ++r) { mi[r] = NEG_BIG; li[r] = 0.0f; }

    const int jend  = qb0 + 127;            // block's last key
    const int qlast = q0 + 15;              // wave's last key

    attn_issue_kv(smem, K, Vt, 0, 0, tid);
    wait_async_and_barrier();

    for (int j0 = 0; j0 <= jend; j0 += 32) {
        const int buf = (j0 >> 5) & 1;
        if (j0 + 32 <= jend) attn_issue_kv(smem, K, Vt, j0 + 32, buf ^ 1, tid);

        if (j0 <= qlast) {                  // wave-local compute guard
            const bf16_t* kt = smK + buf * 2048;   // [32][64]
            const bf16_t* vt = smV + buf * 2048;   // [64][32]

            // ---- scores: S[16x32] = Q(16x64) * K_tile^T ----
            v8f s0 = vzero8(), s1 = vzero8();
            {
                v16bf bk;
                bk = load_b_frag(kt, DHEAD, 0,  0,  lane); s0 = wmma_bf16(aq0, bk, s0);
                bk = load_b_frag(kt, DHEAD, 0,  32, lane); s0 = wmma_bf16(aq1, bk, s0);
                bk = load_b_frag(kt, DHEAD, 16, 0,  lane); s1 = wmma_bf16(aq0, bk, s1);
                bk = load_b_frag(kt, DHEAD, 16, 32, lane); s1 = wmma_bf16(aq1, bk, s1);
            }

            // ---- causal mask + online softmax (row = q0 + r + 8*half) ----
#pragma unroll
            for (int r = 0; r < 8; ++r) {
                const int row = q0 + r + 8 * half;
                const int c0  = j0 + idx;
                const int c1  = c0 + 16;
                float v0 = (c0 <= row) ? s0[r] * SCALE : NEG_BIG;
                float v1 = (c1 <= row) ? s1[r] * SCALE : NEG_BIG;
                float mx = fmaxf(v0, v1);
                mx = fmaxf(mx, __shfl_xor(mx, 8, 16));
                mx = fmaxf(mx, __shfl_xor(mx, 4, 16));
                mx = fmaxf(mx, __shfl_xor(mx, 2, 16));
                mx = fmaxf(mx, __shfl_xor(mx, 1, 16));
                const float mnew  = fmaxf(mi[r], mx);
                const float alpha = __expf(mi[r] - mnew);
                const float p0 = __expf(v0 - mnew);
                const float p1 = __expf(v1 - mnew);
                float rs = p0 + p1;
                rs += __shfl_xor(rs, 8, 16);
                rs += __shfl_xor(rs, 4, 16);
                rs += __shfl_xor(rs, 2, 16);
                rs += __shfl_xor(rs, 1, 16);
                li[r] = li[r] * alpha + rs;
                mi[r] = mnew;
                s0[r] = p0;
                s1[r] = p1;
                o[0][r] *= alpha; o[1][r] *= alpha;
                o[2][r] *= alpha; o[3][r] *= alpha;
            }

            // ---- P: C-layout -> A-layout via per-wave LDS (DS in-order) ----
#pragma unroll
            for (int r = 0; r < 8; ++r) {
                pw[(r + 8 * half) * 32 + idx]      = (bf16_t)s0[r];
                pw[(r + 8 * half) * 32 + 16 + idx] = (bf16_t)s1[r];
            }
            AFrag pf;
            pf.h[0] = *(const v8bf*)&pw[idx * 32 + half * 16];
            pf.h[1] = *(const v8bf*)&pw[idx * 32 + half * 16 + 8];

            // ---- O += P(16x32) * V(32x64) ----
#pragma unroll
            for (int t = 0; t < 4; ++t) {
                v16bf bv = load_b_frag(vt, 32, 16 * t, 0, lane);
                o[t] = wmma_bf16(pf.v, bv, o[t]);
            }
        }

        wait_async_and_barrier();           // uniform across the block
    }

    // ---- epilogue: normalize, merge heads, write bf16 O [B,S,D] ----
    const int b = bh >> 4;
    const int h = bh & 15;
#pragma unroll
    for (int t = 0; t < 4; ++t) {
#pragma unroll
        for (int r = 0; r < 8; ++r) {
            const int s = q0 + r + 8 * half;
            const float val = o[t][r] / li[r];
            Ob[((size_t)(b * SEQ + s)) * DEMB + h * DHEAD + 16 * t + idx] =
                (bf16_t)val;
        }
    }
}

// ---------------------------------------------------------------------------
// 4) Output projection: out = O @ Wo^T, fp32 result. grid = (N/128, M/64).
// ---------------------------------------------------------------------------
__global__ __launch_bounds__(256)
void gemm_out(const bf16_t* __restrict__ Ob,
              const bf16_t* __restrict__ wo,
              float* __restrict__ out) {
    __shared__ __align__(32) char smem[GEMM_LDS_BYTES];
    const int tid  = threadIdx.x;
    const int lane = tid & 31;
    const int w    = tid >> 5;
    const int mblk = blockIdx.y * 64;
    const int nblk = blockIdx.x * 128;

    v8f acc[4];
#pragma unroll
    for (int t = 0; t < 4; ++t) acc[t] = vzero8();

    gemm_core(Ob + (size_t)mblk * DEMB, wo + (size_t)nblk * DEMB,
              acc, smem, tid, lane, w);

    const int half = lane >> 4;
    const int idx  = lane & 15;
    const int m0   = mblk + (w & 3) * 16;
    const int n0   = nblk + (w >> 2) * 64;
#pragma unroll
    for (int t = 0; t < 4; ++t) {
#pragma unroll
        for (int r = 0; r < 8; ++r) {
            const int m = m0 + r + 8 * half;
            const int n = n0 + 16 * t + idx;
            out[(size_t)m * DEMB + n] = acc[t][r];
        }
    }
}

// ---------------------------------------------------------------------------
// Host-side launch
// ---------------------------------------------------------------------------
extern "C" void kernel_launch(void* const* d_in, const int* in_sizes, int n_in,
                              void* d_out, int out_size, void* d_ws, size_t ws_size,
                              hipStream_t stream) {
    const float* x  = (const float*)d_in[0];
    const float* Wq = (const float*)d_in[1];
    const float* Wk = (const float*)d_in[2];
    const float* Wv = (const float*)d_in[3];
    const float* Wo = (const float*)d_in[4];
    float* out = (float*)d_out;

    const size_t nX = (size_t)NTOK * DEMB;
    const size_t nW = (size_t)DEMB * DEMB;
    const size_t nH = (size_t)NTOK * DEMB;

    char* ws = (char*)d_ws;
    bf16_t* xb  = (bf16_t*)ws;                  ws += nX * sizeof(bf16_t);
    bf16_t* wqb = (bf16_t*)ws;                  ws += nW * sizeof(bf16_t);
    bf16_t* wkb = (bf16_t*)ws;                  ws += nW * sizeof(bf16_t);
    bf16_t* wvb = (bf16_t*)ws;                  ws += nW * sizeof(bf16_t);
    bf16_t* wob = (bf16_t*)ws;                  ws += nW * sizeof(bf16_t);
    bf16_t* Qb  = (bf16_t*)ws;                  ws += nH * sizeof(bf16_t);
    bf16_t* Kb  = (bf16_t*)ws;                  ws += nH * sizeof(bf16_t);
    bf16_t* Vtb = (bf16_t*)ws;                  ws += nH * sizeof(bf16_t);
    bf16_t* Ob  = (bf16_t*)ws;                  ws += nH * sizeof(bf16_t);

    {
        const int T = 256;
        cvt_f32_bf16<<<(int)(nX / 4 / T), T, 0, stream>>>(x,  xb,  (int)nX);
        cvt_f32_bf16<<<(int)(nW / 4 / T), T, 0, stream>>>(Wq, wqb, (int)nW);
        cvt_f32_bf16<<<(int)(nW / 4 / T), T, 0, stream>>>(Wk, wkb, (int)nW);
        cvt_f32_bf16<<<(int)(nW / 4 / T), T, 0, stream>>>(Wv, wvb, (int)nW);
        cvt_f32_bf16<<<(int)(nW / 4 / T), T, 0, stream>>>(Wo, wob, (int)nW);
    }

    gemm_qkv<<<dim3(DEMB / 128, NTOK / 64, 3), 256, 0, stream>>>(
        xb, wqb, wkb, wvb, Qb, Kb, Vtb);

    flash_attn<<<dim3(SEQ / 128, BATCH * NHEAD), 256, 0, stream>>>(
        Qb, Kb, Vtb, Ob);

    gemm_out<<<dim3(DEMB / 128, NTOK / 64), 256, 0, stream>>>(Ob, wob, out);
}
